// CFGGNNEncoder_21217138442428
// MI455X (gfx1250) — matrix operand order repaired and checked by reading, not compiled, not run on hardware
//
#include <hip/hip_runtime.h>
#include <hip/hip_bf16.h>

#define DIM 128
#define EPSV 1e-5f
#define SWS 136   // padded LDS row stride (ushorts) for W^T : 272B, 16B-aligned
#define SAS 136   // padded LDS row stride (ushorts) for A tile
#define ROWS 64   // rows per GEMM block

typedef __bf16 bf16_t;
typedef __attribute__((ext_vector_type(16))) __bf16 v16bf;
typedef __attribute__((ext_vector_type(8)))  float  v8f;

union FragU { v16bf v; uint4 u[2]; };

__device__ __forceinline__ unsigned short f2bf(float f) {
    unsigned u = __float_as_uint(f);
    unsigned r = u + 0x7fffu + ((u >> 16) & 1u);   // round-to-nearest-even
    return (unsigned short)(r >> 16);
}

// ---------------- degree / dinv ----------------
__global__ void init_deg_kernel(float* deg, int N) {
    int i = blockIdx.x * blockDim.x + threadIdx.x;
    if (i < N) deg[i] = 1.0f;                      // self loop
}

__global__ void accum_deg_kernel(float* deg, const int* __restrict__ dst, int E) {
    int i = blockIdx.x * blockDim.x + threadIdx.x;
    if (i < E) atomicAdd(&deg[dst[i]], 1.0f);
}

__global__ void finish_dinv_kernel(float* deg, int N) {
    int i = blockIdx.x * blockDim.x + threadIdx.x;
    if (i < N) {
        float d = deg[i];
        deg[i] = (d > 0.0f) ? rsqrtf(d) : 0.0f;
    }
}

// ---------------- WMMA GEMM: tmp = A(N x 128) @ W(128 x 128), bf16 in / f32 acc ----
// Block: 256 threads = 8 waves, 64 rows x 128 cols of output.
// Wave w: row tile (w & 3), column half (w >> 2): 16 rows x 64 cols = 4 WMMA tiles.
__global__ __launch_bounds__(256) void gemm_wmma_kernel(
        const float* __restrict__ A, const float* __restrict__ W,
        float* __restrict__ out, int N) {
    __shared__ __align__(16) unsigned short sWT[DIM * SWS];  // W^T bf16, bank-skewed
    __shared__ __align__(16) unsigned short sA[ROWS * SAS];  // A tile bf16

    const int tid  = threadIdx.x;
    const int row0 = blockIdx.x * ROWS;

    // Load W transposed into LDS: sWT[n*SWS + k] = bf16(W[k*128 + n])
    for (int i = tid; i < DIM * DIM; i += 256) {
        int k = i >> 7, n = i & 127;
        sWT[n * SWS + k] = f2bf(W[i]);
    }
    // Load 64-row A tile into LDS as bf16
    for (int i = tid; i < ROWS * DIM; i += 256) {
        int r = i >> 7, c = i & 127;
        int row = row0 + r;
        sA[r * SAS + c] = (row < N) ? f2bf(A[(size_t)row * DIM + c])
                                    : (unsigned short)0;
    }
    __syncthreads();

    const int wave    = tid >> 5;
    const int lane    = tid & 31;
    const int rowBase = (wave & 3) * 16;   // wave's row tile within block
    const int colBase = (wave >> 2) * 64;  // wave's 64-col half

    // A fragment (contiguous K order, matching B):
    //   lanes 0-15: M=lane,    elems 0-15 = K+0..15
    //   lanes16-31: M=lane-16, elems 0-15 = K+16..31
    const int aRow = rowBase + (lane & 15);
    const int aOff = (lane >> 4) * 16;
    // B fragment (contiguous K order, matching A): lane l: N=l&15, 16 contiguous K
    // starting at kb*32 + (l>>4)*16
    const int bColL = lane & 15;
    const int bOff  = (lane >> 4) * 16;

    v8f c0 = {}, c1 = {}, c2 = {}, c3 = {};
#pragma unroll
    for (int kb = 0; kb < 4; ++kb) {
        const int k0 = kb * 32;
        FragU fa;
        const unsigned short* ap = &sA[aRow * SAS + k0 + aOff];
        fa.u[0] = *(const uint4*)(ap);
        fa.u[1] = *(const uint4*)(ap + 8);

        FragU fb0, fb1, fb2, fb3;
        const unsigned short* bp0 = &sWT[(colBase +  0 + bColL) * SWS + k0 + bOff];
        const unsigned short* bp1 = &sWT[(colBase + 16 + bColL) * SWS + k0 + bOff];
        const unsigned short* bp2 = &sWT[(colBase + 32 + bColL) * SWS + k0 + bOff];
        const unsigned short* bp3 = &sWT[(colBase + 48 + bColL) * SWS + k0 + bOff];
        fb0.u[0] = *(const uint4*)(bp0); fb0.u[1] = *(const uint4*)(bp0 + 8);
        fb1.u[0] = *(const uint4*)(bp1); fb1.u[1] = *(const uint4*)(bp1 + 8);
        fb2.u[0] = *(const uint4*)(bp2); fb2.u[1] = *(const uint4*)(bp2 + 8);
        fb3.u[0] = *(const uint4*)(bp3); fb3.u[1] = *(const uint4*)(bp3 + 8);

        c0 = __builtin_amdgcn_wmma_f32_16x16x32_bf16(false, fa.v, false, fb0.v, (short)0, c0, false, false);
        c1 = __builtin_amdgcn_wmma_f32_16x16x32_bf16(false, fa.v, false, fb1.v, (short)0, c1, false, false);
        c2 = __builtin_amdgcn_wmma_f32_16x16x32_bf16(false, fa.v, false, fb2.v, (short)0, c2, false, false);
        c3 = __builtin_amdgcn_wmma_f32_16x16x32_bf16(false, fa.v, false, fb3.v, (short)0, c3, false, false);
    }

    // C/D layout: VGPR r holds (M=r, N=lane) lanes 0-15; (M=r+8, N=lane-16) else
    const int cCol  = lane & 15;
    const int rBase = (lane >> 4) * 8;
    float* obase = out + (size_t)(row0 + rowBase) * DIM + colBase + cCol;

    if (row0 + ROWS <= N) {          // fast path: no bounds checks
#pragma unroll
        for (int r = 0; r < 8; ++r) {
            size_t ro = (size_t)(rBase + r) * DIM;
            obase[ro +  0] = c0[r];
            obase[ro + 16] = c1[r];
            obase[ro + 32] = c2[r];
            obase[ro + 48] = c3[r];
        }
    } else {                         // tail block only
#pragma unroll
        for (int r = 0; r < 8; ++r) {
            int row = row0 + rowBase + rBase + r;
            if (row < N) {
                size_t ro = (size_t)(rBase + r) * DIM;
                obase[ro +  0] = c0[r];
                obase[ro + 16] = c1[r];
                obase[ro + 32] = c2[r];
                obase[ro + 48] = c3[r];
            }
        }
    }
}

// ---------------- out = b + dinv[n]^2 * tmp[n]   (bias + self-loop term) ---------
__global__ __launch_bounds__(256) void init_out_kernel(
        const float* __restrict__ tmp, const float* __restrict__ dinv,
        const float* __restrict__ b, float* __restrict__ out, int N) {
    int gid = blockIdx.x * blockDim.x + threadIdx.x;
    int n = gid >> 5;
    if (n >= N) return;
    int q = (gid & 31) * 4;
    float w = dinv[n]; w *= w;
    float4 v  = *(const float4*)(tmp + (size_t)n * DIM + q);
    float4 bb = *(const float4*)(b + q);
    float4 r;
    r.x = bb.x + v.x * w; r.y = bb.y + v.y * w;
    r.z = bb.z + v.z * w; r.w = bb.w + v.w * w;
    *(float4*)(out + (size_t)n * DIM + q) = r;
}

// ---------------- edge scatter: out[dst] += dinv[s]*dinv[d] * tmp[src] -----------
__global__ __launch_bounds__(256) void edge_scatter_kernel(
        const float* __restrict__ tmp, const float* __restrict__ dinv,
        const int* __restrict__ src, const int* __restrict__ dst,
        float* __restrict__ out, int E) {
    long long gid = (long long)blockIdx.x * blockDim.x + threadIdx.x;
    int e = (int)(gid >> 5);
    if (e >= E) return;
    int q = ((int)gid & 31) * 4;
    int s = src[e], d = dst[e];
    float w = dinv[s] * dinv[d];
    float4 v = *(const float4*)(tmp + (size_t)s * DIM + q);
    float* o = out + (size_t)d * DIM + q;
    atomicAdd(o + 0, v.x * w);
    atomicAdd(o + 1, v.y * w);
    atomicAdd(o + 2, v.z * w);
    atomicAdd(o + 3, v.w * w);
}

// ---------------- ReLU + LayerNorm (one wave32 per node, float4 per lane) --------
__global__ __launch_bounds__(256) void relu_ln_kernel(
        const float* __restrict__ in, const float* __restrict__ gamma,
        const float* __restrict__ beta, float* __restrict__ out, int N) {
    int gid = blockIdx.x * blockDim.x + threadIdx.x;
    int n = gid >> 5;
    if (n >= N) return;
    int q = (gid & 31) * 4;
    float4 v = *(const float4*)(in + (size_t)n * DIM + q);
    v.x = fmaxf(v.x, 0.0f); v.y = fmaxf(v.y, 0.0f);
    v.z = fmaxf(v.z, 0.0f); v.w = fmaxf(v.w, 0.0f);
    float s  = v.x + v.y + v.z + v.w;
    float s2 = v.x * v.x + v.y * v.y + v.z * v.z + v.w * v.w;
#pragma unroll
    for (int m = 16; m >= 1; m >>= 1) {
        s  += __shfl_xor(s,  m, 32);
        s2 += __shfl_xor(s2, m, 32);
    }
    float mu  = s * (1.0f / DIM);
    float var = s2 * (1.0f / DIM) - mu * mu;
    float inv = rsqrtf(var + EPSV);
    float4 g  = *(const float4*)(gamma + q);
    float4 bb = *(const float4*)(beta + q);
    float4 r;
    r.x = (v.x - mu) * inv * g.x + bb.x;
    r.y = (v.y - mu) * inv * g.y + bb.y;
    r.z = (v.z - mu) * inv * g.z + bb.z;
    r.w = (v.w - mu) * inv * g.w + bb.w;
    *(float4*)(out + (size_t)n * DIM + q) = r;
}

extern "C" void kernel_launch(void* const* d_in, const int* in_sizes, int n_in,
                              void* d_out, int out_size, void* d_ws, size_t ws_size,
                              hipStream_t stream) {
    const float* x     = (const float*)d_in[0];
    const float* W     = (const float*)d_in[1];
    const float* b     = (const float*)d_in[2];
    const float* gamma = (const float*)d_in[3];
    const float* beta  = (const float*)d_in[4];
    const int*   ei    = (const int*)d_in[5];

    const int N = in_sizes[0] / DIM;
    const int L = in_sizes[1] / (DIM * DIM);
    const int E = in_sizes[5] / 2;
    const int* src = ei;
    const int* dst = ei + E;

    char* ws = (char*)d_ws;
    size_t nb   = (((size_t)N * sizeof(float)) + 255) & ~(size_t)255;
    size_t ndb  = (size_t)N * DIM * sizeof(float);
    float* dinv = (float*)ws;
    float* tmp  = (float*)(ws + nb);
    float* aggr = (float*)(ws + nb + ndb);
    float* hbuf = (float*)(ws + nb + 2 * ndb);
    float* outp = (float*)d_out;

    init_deg_kernel   <<<(N + 255) / 256, 256, 0, stream>>>(dinv, N);
    accum_deg_kernel  <<<(E + 255) / 256, 256, 0, stream>>>(dinv, dst, E);
    finish_dinv_kernel<<<(N + 255) / 256, 256, 0, stream>>>(dinv, N);

    const long long nodeThreads = (long long)N * 32;
    const long long edgeThreads = (long long)E * 32;

    const float* hin = x;
    for (int l = 0; l < L; ++l) {
        gemm_wmma_kernel<<<(N + ROWS - 1) / ROWS, 256, 0, stream>>>(
            hin, W + (size_t)l * DIM * DIM, tmp, N);
        init_out_kernel<<<(unsigned)((nodeThreads + 255) / 256), 256, 0, stream>>>(
            tmp, dinv, b + (size_t)l * DIM, aggr, N);
        edge_scatter_kernel<<<(unsigned)((edgeThreads + 255) / 256), 256, 0, stream>>>(
            tmp, dinv, src, dst, aggr, E);
        float* lnout = (l == L - 1) ? outp : hbuf;
        relu_ln_kernel<<<(unsigned)((nodeThreads + 255) / 256), 256, 0, stream>>>(
            aggr, gamma + (size_t)l * DIM, beta + (size_t)l * DIM, lnout, N);
        hin = hbuf;
    }
}
